// MultiGraph_53300544143826
// MI455X (gfx1250) — matrix operand order repaired and checked by reading, not compiled, not run on hardware
//
#include <hip/hip_runtime.h>

typedef __attribute__((ext_vector_type(16))) __bf16 v16bf;
typedef __attribute__((ext_vector_type(8)))  float  v8f;
typedef __attribute__((ext_vector_type(8)))  unsigned short ush8;
typedef __attribute__((ext_vector_type(4)))  int    v4i;

#define B_SZ   128
#define L_SZ   336
#define DIM_SZ 862
#define H_SZ   8
#define D_SZ   128
#define P_SZ   336
#define HD_SZ  1024
#define LD_SZ  (L_SZ * D_SZ)     /* 43008 */
#define BHLD   (H_SZ * LD_SZ)    /* 344064 elements per batch */

// ---- CDNA5 async global->LDS copy (probe via __has_builtin; safe fallback) ----
#if defined(__has_builtin)
#  if __has_builtin(__builtin_amdgcn_global_load_async_to_lds_b128)
#    define HAVE_ASYNC_LDS 1
#  endif
#  if __has_builtin(__builtin_amdgcn_s_wait_asynccnt)
#    define HAVE_WAIT_ASYNC 1
#  endif
#endif

// pointer-to-AS(1) int4 and pointer-to-AS(3) int4
typedef __attribute__((address_space(1))) v4i* glb_v4i_p;
typedef __attribute__((address_space(3))) v4i* lds_v4i_p;

static __device__ __forceinline__ void async_copy16(unsigned short* lds_dst,
                                                    const unsigned short* g_src) {
#if defined(HAVE_ASYNC_LDS)
  __builtin_amdgcn_global_load_async_to_lds_b128(
      (glb_v4i_p)g_src, (lds_v4i_p)lds_dst, 0, 0);
#else
  *(ush8*)lds_dst = *(const ush8*)g_src;
#endif
}

static __device__ __forceinline__ void wait_async_barrier() {
#if defined(HAVE_ASYNC_LDS)
#  if defined(HAVE_WAIT_ASYNC)
  __builtin_amdgcn_s_wait_asynccnt(0);
#  else
  asm volatile("s_wait_asynccnt 0" ::: "memory");
#  endif
#endif
  __syncthreads();
}

static __device__ __forceinline__ unsigned short f2bf(float f) {
  unsigned int u = __builtin_bit_cast(unsigned int, f);
  u += 0x7FFFu + ((u >> 16) & 1u);          // round-to-nearest-even
  return (unsigned short)(u >> 16);
}
static __device__ __forceinline__ float bf2f(unsigned short h) {
  unsigned int u = ((unsigned int)h) << 16;
  return __builtin_bit_cast(float, u);
}

// ---------------- f32 -> bf16 with zero padding ----------------
__global__ void convert_pad_kernel(const float* __restrict__ src,
                                   unsigned short* __restrict__ dst,
                                   int rows, int cols, int colsPad,
                                   long long total) {
  long long idx = (long long)blockIdx.x * blockDim.x + threadIdx.x;
  if (idx >= total) return;
  int r = (int)(idx / colsPad);
  int c = (int)(idx % colsPad);
  unsigned short v = 0;
  if (r < rows && c < cols) v = f2bf(src[(long long)r * cols + c]);
  dst[idx] = v;
}

// ---------------- bf16 WMMA GEMM: C[M,N] = A[M,K] * B + bias ----------------
// B_TRANS=1 : B stored [N,K] row-major (weight-transposed, like torch Linear)
// B_TRANS=0 : B stored [K,N] row-major (transposed into LDS during staging)
// BIAS_MODE : 0 none, 1 bias[n], 2 bias[m]
// OUT_MODE  : 0 bf16 flat [M,ldc], 1 bf16 permuted [B,H,L,D], 2 f32 flat
#define TM 128
#define TN 128
#define TK 32
#define LSTR 40   /* LDS row stride in halves: 80B, 16B aligned, bank-spread */

template <int BIAS_MODE, int B_TRANS, int OUT_MODE>
__global__ __launch_bounds__(256) void gemm_bf16_wmma(
    const unsigned short* __restrict__ A, int lda, long long strideA,
    const unsigned short* __restrict__ Bm, int ldb, long long strideB,
    const float* __restrict__ bias,
    void* __restrict__ Cp, int ldc, long long strideC,
    int Kpad, int Kval, int Mval, int Nval) {
  __shared__ __align__(16) unsigned short lA[2][TM * LSTR];
  __shared__ __align__(16) unsigned short lB[2][TN * LSTR];

  const int z = blockIdx.z;
  A  += (long long)z * strideA;
  Bm += (long long)z * strideB;

  const int mblk = blockIdx.y * TM;
  const int nblk = blockIdx.x * TN;
  const int tid  = threadIdx.x;
  const int lane = tid & 31;
  const int wave = tid >> 5;
  const int wm = wave & 3;    // 4 waves along M -> 32 rows each
  const int wn = wave >> 2;   // 2 waves along N -> 64 cols each
  const int lr  = lane & 15;
  const int ksA = (lane & 16) ? 16 : 0;   // A frag: K {0..15} | {16..31}
  const int ksB = (lane & 16) ? 16 : 0;   // B frag: K {0..15} | {16..31}

  auto stageA = [&](int buf, int kt) {
#pragma unroll
    for (int it = 0; it < 2; ++it) {
      int s  = tid + it * 256;          // 512 x 16B segments
      int m  = s >> 2;
      int k0 = (s & 3) * 8;
      async_copy16(&lA[buf][m * LSTR + k0],
                   A + (long long)(mblk + m) * lda + (kt + k0));
    }
  };
  auto stageB = [&](int buf, int kt) {
    if (B_TRANS) {
#pragma unroll
      for (int it = 0; it < 2; ++it) {
        int s  = tid + it * 256;
        int n  = s >> 2;
        int k0 = (s & 3) * 8;
        async_copy16(&lB[buf][n * LSTR + k0],
                     Bm + (long long)(nblk + n) * ldb + (kt + k0));
      }
    } else if (kt + TK <= Kval) {       // uniform: full tile, no guards
#pragma unroll
      for (int it = 0; it < 2; ++it) {
        int s  = tid + it * 256;
        int k  = s >> 4;
        int n0 = (s & 15) * 8;
        ush8 val = *(const ush8*)(Bm + (long long)(kt + k) * ldb + (nblk + n0));
#pragma unroll
        for (int j = 0; j < 8; ++j) lB[buf][(n0 + j) * LSTR + k] = val[j];
      }
    } else {                            // uniform: K-tail tile, zero-fill
#pragma unroll
      for (int it = 0; it < 2; ++it) {
        int s  = tid + it * 256;
        int k  = s >> 4;
        int n0 = (s & 15) * 8;
        ush8 val;
#pragma unroll
        for (int j = 0; j < 8; ++j) val[j] = 0;
        if (kt + k < Kval)
          val = *(const ush8*)(Bm + (long long)(kt + k) * ldb + (nblk + n0));
#pragma unroll
        for (int j = 0; j < 8; ++j) lB[buf][(n0 + j) * LSTR + k] = val[j];
      }
    }
  };

  v8f acc[2][4];
#pragma unroll
  for (int mi = 0; mi < 2; ++mi)
#pragma unroll
    for (int ni = 0; ni < 4; ++ni)
#pragma unroll
      for (int v = 0; v < 8; ++v) acc[mi][ni][v] = 0.f;

  // prologue: stage first K-tile into buffer 0
  stageA(0, 0);
  stageB(0, 0);

  int buf = 0;
  for (int kt = 0; kt < Kpad; kt += TK, buf ^= 1) {
    wait_async_barrier();               // current buf visible to all waves
    if (kt + TK < Kpad) {               // stage next tile into other buffer
      stageA(buf ^ 1, kt + TK);
      stageB(buf ^ 1, kt + TK);
      if (kt + 2 * TK < Kpad) {
        // one 64B-line prefetch per thread; wave-aligned split (0-3:A, 4-7:B)
        const unsigned short* pf;
        if (B_TRANS) {
          pf = (tid < TM)
                   ? A  + (long long)(mblk + tid) * lda + (kt + 2 * TK)
                   : Bm + (long long)(nblk + (tid - TM)) * ldb + (kt + 2 * TK);
        } else {
          if (tid < TM) {
            pf = A + (long long)(mblk + tid) * lda + (kt + 2 * TK);
          } else {
            int t = tid - TM;           // 128 threads cover 32 rows x 4 lines
            pf = Bm + (long long)(kt + 2 * TK + (t >> 2)) * ldb +
                 (nblk + (t & 3) * 32);
          }
        }
        __builtin_prefetch(pf, 0, 1);
      }
    }
    const unsigned short* lAp = lA[buf];
    const unsigned short* lBp = lB[buf];

    union FU { v16bf v; ush8 u[2]; };
    FU af[2], bfg[4];
#pragma unroll
    for (int mi = 0; mi < 2; ++mi) {
      int row = wm * 32 + mi * 16 + lr;
      af[mi].u[0] = *(const ush8*)(lAp + row * LSTR + ksA);
      af[mi].u[1] = *(const ush8*)(lAp + row * LSTR + ksA + 8);
    }
#pragma unroll
    for (int ni = 0; ni < 4; ++ni) {
      int row = wn * 64 + ni * 16 + lr;
      bfg[ni].u[0] = *(const ush8*)(lBp + row * LSTR + ksB);
      bfg[ni].u[1] = *(const ush8*)(lBp + row * LSTR + ksB + 8);
    }
#pragma unroll
    for (int mi = 0; mi < 2; ++mi)
#pragma unroll
      for (int ni = 0; ni < 4; ++ni)
        acc[mi][ni] = __builtin_amdgcn_wmma_f32_16x16x32_bf16(
            false, af[mi].v, false, bfg[ni].v, (short)0, acc[mi][ni],
            false, false);
  }

  // ---- epilogue: bias + store ----
  const int lm = (lane >> 4) * 8;
#pragma unroll
  for (int mi = 0; mi < 2; ++mi)
#pragma unroll
    for (int ni = 0; ni < 4; ++ni)
#pragma unroll
      for (int v = 0; v < 8; ++v) {
        int m = mblk + wm * 32 + mi * 16 + lm + v;
        int n = nblk + wn * 64 + ni * 16 + lr;
        if (m < Mval && n < Nval) {
          float c = acc[mi][ni][v];
          if (BIAS_MODE == 1) c += bias[n];
          if (BIAS_MODE == 2) c += bias[m];
          if (OUT_MODE == 0) {
            unsigned short* C = (unsigned short*)Cp + (long long)z * strideC;
            C[(long long)m * ldc + n] = f2bf(c);
          } else if (OUT_MODE == 1) {
            // m=(b,l), n=(h,d): store permuted [B,H,L,D] bf16
            int bb = m / L_SZ, l = m - bb * L_SZ;
            int h = n >> 7, dd = n & 127;
            unsigned short* C = (unsigned short*)Cp;
            C[((long long)bb * H_SZ + h) * LD_SZ + (long long)l * D_SZ + dd] =
                f2bf(c);
          } else {
            float* C = (float*)Cp + (long long)z * strideC;
            C[(long long)m * ldc + n] = c;
          }
        }
      }
}

// ---------------- per-batch 8x8 gram matrix + softmax ----------------
__global__ __launch_bounds__(256) void gram_softmax_kernel(
    const unsigned short* __restrict__ mfP, float* __restrict__ att) {
  const int b = blockIdx.x, tid = threadIdx.x;
  const unsigned short* base = mfP + (long long)b * BHLD;
  float acc[36];
#pragma unroll
  for (int p = 0; p < 36; ++p) acc[p] = 0.f;

  for (int i0 = tid * 8; i0 < LD_SZ; i0 += 256 * 8) {   // 43008/2048 = 21 iters
    float v[H_SZ][8];
#pragma unroll
    for (int h = 0; h < H_SZ; ++h) {
      ush8 x = *(const ush8*)(base + (long long)h * LD_SZ + i0);
#pragma unroll
      for (int j = 0; j < 8; ++j) v[h][j] = bf2f(x[j]);
    }
#pragma unroll
    for (int s = 0; s < H_SZ; ++s)
#pragma unroll
      for (int h = s; h < H_SZ; ++h) {
        float d = 0.f;
#pragma unroll
        for (int j = 0; j < 8; ++j) d += v[s][j] * v[h][j];
        acc[s * H_SZ + h - (s * (s + 1)) / 2] += d;   // upper-triangle index
      }
  }
#pragma unroll
  for (int p = 0; p < 36; ++p)
    for (int off = 16; off > 0; off >>= 1)
      acc[p] += __shfl_xor(acc[p], off, 32);

  __shared__ float red[36 * 8];
  __shared__ float g[64];
  const int wv = tid >> 5, lane = tid & 31;
  if (lane == 0)
    for (int p = 0; p < 36; ++p) red[p * 8 + wv] = acc[p];
  __syncthreads();
  if (tid < 36) {
    float s = 0.f;
    for (int w = 0; w < 8; ++w) s += red[tid * 8 + w];
    int p = tid, si = 0;
    while (p >= H_SZ - si) { p -= H_SZ - si; ++si; }
    int hi = si + p;
    g[si * 8 + hi] = s;
    g[hi * 8 + si] = s;   // symmetric gram
  }
  __syncthreads();
  if (tid < 8) {
    float mx = -3.4e38f;
    for (int h = 0; h < 8; ++h) mx = fmaxf(mx, g[tid * 8 + h]);
    float e[8], sum = 0.f;
    for (int h = 0; h < 8; ++h) { e[h] = __expf(g[tid * 8 + h] - mx); sum += e[h]; }
    float inv = 1.f / sum;
    for (int h = 0; h < 8; ++h) att[(long long)b * 64 + tid * 8 + h] = e[h] * inv;
  }
}

// ---------------- new_mf = att @ mf (flat layout reinterpretation) ----------
__global__ __launch_bounds__(256) void apply_att_kernel(
    const unsigned short* __restrict__ mfP, const float* __restrict__ att,
    unsigned short* __restrict__ nm) {
  const int b = blockIdx.y;
  __shared__ float a[64];
  if (threadIdx.x < 64) a[threadIdx.x] = att[(long long)b * 64 + threadIdx.x];
  __syncthreads();
  const long long i0 = ((long long)blockIdx.x * 256 + threadIdx.x) * 8;
  const unsigned short* base = mfP + (long long)b * BHLD;
  unsigned short* ob = nm + (long long)b * BHLD;
  float v[H_SZ][8];
#pragma unroll
  for (int h = 0; h < H_SZ; ++h) {
    ush8 x = *(const ush8*)(base + (long long)h * LD_SZ + i0);
#pragma unroll
    for (int j = 0; j < 8; ++j) v[h][j] = bf2f(x[j]);
  }
#pragma unroll
  for (int s = 0; s < H_SZ; ++s) {
    float o[8];
#pragma unroll
    for (int j = 0; j < 8; ++j) o[j] = 0.f;
#pragma unroll
    for (int h = 0; h < H_SZ; ++h) {
      float w = a[s * 8 + h];
#pragma unroll
      for (int j = 0; j < 8; ++j) o[j] += w * v[h][j];
    }
    ush8 wvv;
#pragma unroll
    for (int j = 0; j < 8; ++j) wvv[j] = f2bf(o[j]);
    *(ush8*)(ob + (long long)s * LD_SZ + i0) = wvv;
  }
}

extern "C" void kernel_launch(void* const* d_in, const int* in_sizes, int n_in,
                              void* d_out, int out_size, void* d_ws, size_t ws_size,
                              hipStream_t stream) {
  (void)in_sizes; (void)n_in; (void)out_size; (void)ws_size;
  const float* x  = (const float*)d_in[0];
  const float* Wp = (const float*)d_in[1];
  const float* bp = (const float*)d_in[2];
  const float* Wl = (const float*)d_in[3];
  const float* bl = (const float*)d_in[4];
  const float* Wo = (const float*)d_in[5];
  const float* bo = (const float*)d_in[6];
  float* out = (float*)d_out;

  // workspace carve-up (all offsets 256B aligned)
  char* w = (char*)d_ws;
  unsigned short* xb  = (unsigned short*)(w + 0);            // 43008 x 864 bf16
  unsigned short* Wpb = (unsigned short*)(w + 74317824ll);   // 1024 x 864
  unsigned short* Wlb = (unsigned short*)(w + 76087296ll);   // 384 x 352
  unsigned short* Wob = (unsigned short*)(w + 76357632ll);   // 896 x 1024
  unsigned short* mfP = (unsigned short*)(w + 78192640ll);   // [B,H,L,D] bf16
  float*          att = (float*)(w + 166273024ll);           // [B,8,8] f32
  unsigned short* nm  = (unsigned short*)(w + 166305792ll);  // [B,344064] bf16
  unsigned short* ob  = (unsigned short*)(w + 254386176ll);  // [B,336,1024] bf16

  // f32 -> bf16 padded conversions
  { long long t = 43008ll * 864;
    convert_pad_kernel<<<dim3((unsigned)((t + 255) / 256)), 256, 0, stream>>>(
        x, xb, 43008, 862, 864, t); }
  { long long t = 1024ll * 864;
    convert_pad_kernel<<<dim3((unsigned)((t + 255) / 256)), 256, 0, stream>>>(
        Wp, Wpb, 1024, 862, 864, t); }
  { long long t = 384ll * 352;
    convert_pad_kernel<<<dim3((unsigned)((t + 255) / 256)), 256, 0, stream>>>(
        Wl, Wlb, 336, 336, 352, t); }
  { long long t = 896ll * 1024;
    convert_pad_kernel<<<dim3((unsigned)((t + 255) / 256)), 256, 0, stream>>>(
        Wo, Wob, 862, 1024, 1024, t); }

  // GEMM1: mf = x @ Wp^T + bp, stored permuted [B,H,L,D]
  gemm_bf16_wmma<1, 1, 1><<<dim3(8, 336, 1), 256, 0, stream>>>(
      xb, 864, 0ll, Wpb, 864, 0ll, bp, (void*)mfP, 0, 0ll,
      864, 864, 43008, 1024);

  // att = softmax(mf @ mf^T) per batch
  gram_softmax_kernel<<<dim3(128), 256, 0, stream>>>(mfP, att);

  // new_mf = att @ mf  (flat buffer doubles as [B,L,HD])
  apply_att_kernel<<<dim3(21, 128), 256, 0, stream>>>(mfP, att, nm);

  // GEMM2 (batched): out_b = Wl @ new_mf_b + bl[p]
  gemm_bf16_wmma<2, 0, 0><<<dim3(8, 3, 128), 256, 0, stream>>>(
      Wlb, 352, 0ll, nm, 1024, 344064ll, bl, (void*)ob, 1024, 344064ll,
      352, 336, 336, 1024);

  // GEMM3: final = out @ Wo^T + bo  (f32 output)
  gemm_bf16_wmma<1, 1, 2><<<dim3(7, 336, 1), 256, 0, stream>>>(
      ob, 1024, 0ll, Wob, 1024, 0ll, bo, (void*)out, 862, 0ll,
      1024, 1024, 43008, 862);
}
